// PermutationNet_90254442758407
// MI455X (gfx1250) — matrix operand order repaired and checked by reading, not compiled, not run on hardware
//
#include <hip/hip_runtime.h>

// PermutationNet on MI455X (gfx1250), wave32 + WMMA f16->f32.
//
// Pipeline per permutation layer:
//   K1: A1/A2 = h @ W_first_half.T, h @ W_second_half.T (real h x complex W)
//       -> 4 small f32 GEMMs (2048 x f x 256), VALU.
//   K2: workgroup per (b,i): build u=relu(A1r[b,j]+A2r[b,i]+b1r),
//       v=relu(imag) in LDS (f16), then complex 256->256 GEMM with
//       v_wmma_f32_16x16x32_f16 (Zr = u*Wr - v*Wi, Zi = u*Wi + v*Wr),
//       epilogue: |Z| + mean over j (layers 0/1) or relu + complex
//       256->1 dot + |.| + mean over j (output subnet).

#define KSTRIDE_ 264   // padded LDS row stride (f16): 528B/row -> conflict-free b128

typedef _Float16 f16;
typedef __attribute__((ext_vector_type(16))) _Float16 v16h;
typedef __attribute__((ext_vector_type(8)))  _Float16 v8h;
typedef __attribute__((ext_vector_type(8)))  float    v8f;

static __device__ inline v16h cat16(v8h lo, v8h hi) {
  v16h r;
#pragma unroll
  for (int t = 0; t < 8; ++t) { r[t] = lo[t]; r[t + 8] = hi[t]; }
  return r;
}

// ---- K0: convert the three second-linear complex64 weights to split f16 Wr/Wi
__global__ __launch_bounds__(256)
void convert_w_kernel(const float* __restrict__ W0, const float* __restrict__ W1,
                      const float* __restrict__ W2, f16* __restrict__ dst) {
  int blk   = blockIdx.x;        // 0..767
  int layer = blk >> 8;
  int row   = blk & 255;
  int c     = threadIdx.x;       // 0..255
  const float* W = (layer == 0) ? W0 : (layer == 1) ? W1 : W2;
  float re = W[(row * 256 + c) * 2 + 0];
  float im = W[(row * 256 + c) * 2 + 1];
  f16* dr = dst + (size_t)layer * 2 * 65536;
  f16* di = dr + 65536;
  dr[row * 256 + c] = (f16)re;
  di[row * 256 + c] = (f16)im;
}

// ---- K1: A1{r,i}[m,c] = sum_f h[m,f]*W[c, f], A2{r,i}[m,c] = sum_f h[m,f]*W[c, fin+f]
__global__ __launch_bounds__(256)
void stage_a_kernel(const float* __restrict__ h, const float* __restrict__ W0, int fin,
                    float* __restrict__ A1r, float* __restrict__ A1i,
                    float* __restrict__ A2r, float* __restrict__ A2i) {
  __shared__ float sh[256];
  int m = blockIdx.x;      // 0..2047  (b*64+n)
  int c = threadIdx.x;     // 0..255   output channel
  if (c < fin) sh[c] = h[m * fin + c];
  __syncthreads();
  const float2* Wc = (const float2*)W0;  // complex elems, row stride 2*fin
  float a1r = 0.f, a1i = 0.f, a2r = 0.f, a2i = 0.f;
  for (int f = 0; f < fin; ++f) {
    float hv = sh[f];
    float2 w1 = Wc[c * 2 * fin + f];
    float2 w2 = Wc[c * 2 * fin + fin + f];
    a1r += hv * w1.x; a1i += hv * w1.y;
    a2r += hv * w2.x; a2i += hv * w2.y;
  }
  int o = m * 256 + c;
  A1r[o] = a1r; A1i[o] = a1i; A2r[o] = a2r; A2i[o] = a2i;
}

// ---- K2: fused pair-construction + complex WMMA GEMM + epilogue.
// MODE 0: out = h_next (2048 x 256): mean_j |Z|.
// MODE 1: out (2048): relu(Z) -> complex dot with w3 -> +b3 -> |.| -> mean_j.
template <int MODE>
__global__ __launch_bounds__(512)
void pair_gemm_kernel(const float* __restrict__ A1r, const float* __restrict__ A1i,
                      const float* __restrict__ A2r, const float* __restrict__ A2i,
                      const float* __restrict__ bias1,   // c64 interleaved, 256
                      const f16* __restrict__ Wr, const f16* __restrict__ Wi,
                      const float* __restrict__ bias2,   // c64 interleaved, 256
                      const float* __restrict__ w3,      // MODE1: c64 interleaved, 256
                      const float* __restrict__ b3,      // MODE1: c64 interleaved, 1
                      float* __restrict__ out) {
  __shared__ __align__(16) f16 su[32 * KSTRIDE_];
  __shared__ __align__(16) f16 sv[32 * KSTRIDE_];
  __shared__ float sredR[64];
  __shared__ float sredI[64];

  int blk  = blockIdx.x;        // 0..2047 -> (b,i)
  int b    = blk >> 6;
  int i    = blk & 63;
  int tid  = threadIdx.x;
  int wave = tid >> 5;          // 0..15 -> column tile
  int lane = tid & 31;
  int n    = lane & 15;         // A row / B col / D col within tile
  int hi   = lane >> 4;
  int ng   = wave * 16 + n;     // global output channel

  if (MODE == 1) {
    if (tid < 64) { sredR[tid] = b3[0]; sredI[tid] = b3[1]; }
  }

  float b2r = bias2[2 * ng];
  float b2i = bias2[2 * ng + 1];
  float w3r = 0.f, w3i = 0.f;
  if (MODE == 1) { w3r = w3[2 * ng]; w3i = w3[2 * ng + 1]; }

  const int rowA2 = (b * 64 + i) * 256;
  float csum = 0.f;

  for (int halfblk = 0; halfblk < 2; ++halfblk) {
    __syncthreads();   // previous tiles done before LDS overwrite
    // Build u, v for j in [32*halfblk, 32*halfblk+32): 512 thr x 16 elems
#pragma unroll
    for (int t = 0; t < 16; ++t) {
      int idx = tid + t * 512;            // 0..8191
      int r = idx >> 8;                   // local row 0..31
      int k = idx & 255;
      int j = halfblk * 32 + r;
      int m1 = (b * 64 + j) * 256 + k;
      float ur = A1r[m1] + A2r[rowA2 + k] + bias1[2 * k];
      float vi = A1i[m1] + A2i[rowA2 + k] + bias1[2 * k + 1];
      su[r * KSTRIDE_ + k] = (f16)fmaxf(ur, 0.f);
      sv[r * KSTRIDE_ + k] = (f16)fmaxf(vi, 0.f);
    }
    __syncthreads();

    for (int rt = 0; rt < 2; ++rt) {
      int j0 = rt * 16;   // local LDS row base of this 16-row tile
      v8f P1 = {}, P2 = {}, P3 = {}, P4 = {};
#pragma unroll
      for (int kc = 0; kc < 256; kc += 32) {
        // A fragment: lane holds row n; k = kc+16*hi..+15 contiguous
        const f16* ub = &su[(j0 + n) * KSTRIDE_ + kc + 16 * hi];
        const f16* vb = &sv[(j0 + n) * KSTRIDE_ + kc + 16 * hi];
        v16h au = cat16(*(const v8h*)ub, *(const v8h*)(ub + 8));
        v16h av = cat16(*(const v8h*)vb, *(const v8h*)(vb + 8));
        // B fragment: lane holds col n; k = kc+16*hi..+15 contiguous (row-major W[n][k])
        v16h br = *(const v16h*)&Wr[ng * 256 + kc + 16 * hi];
        v16h bi = *(const v16h*)&Wi[ng * 256 + kc + 16 * hi];
        P1 = __builtin_amdgcn_wmma_f32_16x16x32_f16(false, au, false, br, (short)0, P1, false, false);
        P2 = __builtin_amdgcn_wmma_f32_16x16x32_f16(false, av, false, bi, (short)0, P2, false, false);
        P3 = __builtin_amdgcn_wmma_f32_16x16x32_f16(false, au, false, bi, (short)0, P3, false, false);
        P4 = __builtin_amdgcn_wmma_f32_16x16x32_f16(false, av, false, br, (short)0, P4, false, false);
      }
#pragma unroll
      for (int g = 0; g < 8; ++g) {        // D row = g + 8*hi, col = n
        float zr = P1[g] - P2[g] + b2r;
        float zi = P3[g] + P4[g] + b2i;
        if (MODE == 0) {
          csum += sqrtf(zr * zr + zi * zi);          // |Z|, summed over rows (j)
        } else {
          float u2 = fmaxf(zr, 0.f);
          float v2 = fmaxf(zi, 0.f);
          float pr = u2 * w3r - v2 * w3i;
          float pi = u2 * w3i + v2 * w3r;
          int jrow = halfblk * 32 + j0 + g + 8 * hi;
          atomicAdd(&sredR[jrow], pr);
          atomicAdd(&sredI[jrow], pi);
        }
      }
    }
  }

  if (MODE == 0) {
    csum += __shfl_xor(csum, 16, 32);      // combine row halves (same col n)
    if (lane < 16) out[(b * 64 + i) * 256 + ng] = csum * (1.0f / 64.0f);
  } else {
    __syncthreads();
    if (tid == 0) {
      float s = 0.f;
      for (int j = 0; j < 64; ++j)
        s += sqrtf(sredR[j] * sredR[j] + sredI[j] * sredI[j]);
      out[b * 64 + i] = s * (1.0f / 64.0f);
    }
  }
}

extern "C" void kernel_launch(void* const* d_in, const int* in_sizes, int n_in,
                              void* d_out, int out_size, void* d_ws, size_t ws_size,
                              hipStream_t stream) {
  (void)in_sizes; (void)n_in; (void)out_size; (void)ws_size;
  const float* x    = (const float*)d_in[0];
  const float* p0W0 = (const float*)d_in[1];
  const float* p0b0 = (const float*)d_in[2];
  const float* p0W1 = (const float*)d_in[3];
  const float* p0b1 = (const float*)d_in[4];
  const float* p1W0 = (const float*)d_in[5];
  const float* p1b0 = (const float*)d_in[6];
  const float* p1W1 = (const float*)d_in[7];
  const float* p1b1 = (const float*)d_in[8];
  const float* p2W0 = (const float*)d_in[9];
  const float* p2b0 = (const float*)d_in[10];
  const float* p2W1 = (const float*)d_in[11];
  const float* p2b1 = (const float*)d_in[12];
  const float* p2W2 = (const float*)d_in[13];
  const float* p2b2 = (const float*)d_in[14];
  float* out = (float*)d_out;

  float* ws = (float*)d_ws;
  const size_t SEG = 2048 * 256;            // 524288 floats
  float* A1r = ws;
  float* A1i = A1r + SEG;
  float* A2r = A1i + SEG;
  float* A2i = A2r + SEG;
  float* h1  = A2i + SEG;
  float* h2  = h1 + SEG;
  f16*   Wf  = (f16*)(h2 + SEG);            // 6 x 65536 halfs
  f16* W0r = Wf;            f16* W0i = W0r + 65536;
  f16* W1r = W0i + 65536;   f16* W1i = W1r + 65536;
  f16* W2r = W1i + 65536;   f16* W2i = W2r + 65536;

  convert_w_kernel<<<768, 256, 0, stream>>>(p0W1, p1W1, p2W1, Wf);

  // ---- permutation layer 0 (f_in = 128)
  stage_a_kernel<<<2048, 256, 0, stream>>>(x, p0W0, 128, A1r, A1i, A2r, A2i);
  pair_gemm_kernel<0><<<2048, 512, 0, stream>>>(A1r, A1i, A2r, A2i, p0b0,
                                                W0r, W0i, p0b1, nullptr, nullptr, h1);
  // ---- permutation layer 1 (f_in = 256)
  stage_a_kernel<<<2048, 256, 0, stream>>>(h1, p1W0, 256, A1r, A1i, A2r, A2i);
  pair_gemm_kernel<0><<<2048, 512, 0, stream>>>(A1r, A1i, A2r, A2i, p1b0,
                                                W1r, W1i, p1b1, nullptr, nullptr, h2);
  // ---- output permutation layer (f_in = 256, extra relu + 256->1 complex dot)
  stage_a_kernel<<<2048, 256, 0, stream>>>(h2, p2W0, 256, A1r, A1i, A2r, A2i);
  pair_gemm_kernel<1><<<2048, 512, 0, stream>>>(A1r, A1i, A2r, A2i, p2b0,
                                                W2r, W2i, p2b1, p2W2, p2b2, out);
}
